// AMIR_11098195493476
// MI455X (gfx1250) — matrix operand order repaired and checked by reading, not compiled, not run on hardware
//
#include <hip/hip_runtime.h>
#include <hip/hip_bf16.h>

// ---------------------------------------------------------------- constants
#define NTOK   32768
#define DDIM   192
#define HDIM   384
#define NEXP   8
#define TOPK   2
#define EPSV   2.220446049250313e-16f

#define MT     32            // token tile (slots) per block
#define LDX    200           // LDS stride (halves): 400B row (16B aligned), 100 dwords (conflict-free)
#define LDB    200
#define LDH    386           // H tile stride (193 dwords/row, conflict-free)

#define WELEM  (NEXP*DDIM*HDIM)          // 589824
#define XELEM  (NTOK*DDIM)               // 6291456
#define SLOT_CAP (2*NTOK + 256)
#define MAX_TILES ((2*NTOK)/MT + NEXP)   // 2056

typedef _Float16 half_t;
typedef __attribute__((ext_vector_type(16))) _Float16 v16h;
typedef __attribute__((ext_vector_type(2)))  _Float16 h2;
typedef __attribute__((ext_vector_type(8)))  float    v8f;

// ---------------------------------------------------------------- ws layout (bytes)
#define OFF_COUNTS   0
#define OFF_OFFSETS  64
#define OFF_CURSOR   128
#define OFF_TOPIDX   256                          // N*2 ints   = 262144
#define OFF_TOPGATE  (OFF_TOPIDX  + 262144)       // N*2 floats = 262144
#define OFF_SLOTROW  (OFF_TOPGATE + 262144)       // SLOT_CAP ints   = 263168
#define OFF_SLOTGATE (OFF_SLOTROW + 263168)       // SLOT_CAP floats = 263168
#define OFF_W1T      (OFF_SLOTGATE + 263168)      // WELEM halves = 1179648  (transposed [e][n][k])
#define OFF_W2T      (OFF_W1T + 1179648)          // WELEM halves = 1179648  (transposed [e][n][k])
#define OFF_XH       (OFF_W2T + 1179648)          // XELEM halves = 12582912

// ---------------------------------------------------------------- async copy helpers
// VGLOBAL async encoding (ISA 15.18.3 op 98): VDST = LDS byte address VGPR,
// VADDR = 64-bit global address, SADDR = off.  Tracked by ASYNCcnt.
__device__ inline void async_copy16(const half_t* g, half_t* l)
{
    unsigned loff = (unsigned)(unsigned long long)
        (__attribute__((address_space(3))) half_t*)l;     // LDS byte offset
    asm volatile("global_load_async_to_lds_b128 %0, %1, off"
                 :: "v"(loff), "v"((unsigned long long)g) : "memory");
}

__device__ inline void async_wait0()
{
#if __has_builtin(__builtin_amdgcn_s_wait_asynccnt)
    __builtin_amdgcn_s_wait_asynccnt(0);
#else
    asm volatile("s_wait_asynccnt 0x0" ::: "memory");
#endif
}

// ================================================================ gating
__global__ __launch_bounds__(256) void moe_gate_kernel(
    const float* __restrict__ x, const float* __restrict__ emb,
    const float* __restrict__ w_gate,
    int* __restrict__ top_idx, float* __restrict__ top_gate,
    int* __restrict__ counts)
{
    __shared__ float wg[2 * DDIM * NEXP];     // 12 KB
    const int tid = threadIdx.x;
    for (int i = tid; i < 2 * DDIM * NEXP; i += 256) wg[i] = w_gate[i];
    __syncthreads();

    const int row = blockIdx.x * 256 + tid;
    float acc[NEXP];
#pragma unroll
    for (int e = 0; e < NEXP; ++e) acc[e] = 0.f;

    const float4* xv = (const float4*)(x + (size_t)row * DDIM);
    const float4* ev = (const float4*)(emb + (size_t)row * DDIM);
#pragma unroll 4
    for (int j = 0; j < DDIM / 4; ++j) {
        float4 v = xv[j];
        float c[4] = {v.x, v.y, v.z, v.w};
#pragma unroll
        for (int q = 0; q < 4; ++q) {
            const float* wrow = &wg[(4 * j + q) * NEXP];
#pragma unroll
            for (int e = 0; e < NEXP; ++e) acc[e] += c[q] * wrow[e];
        }
    }
#pragma unroll 4
    for (int j = 0; j < DDIM / 4; ++j) {
        float4 v = ev[j];
        float c[4] = {v.x, v.y, v.z, v.w};
#pragma unroll
        for (int q = 0; q < 4; ++q) {
            const float* wrow = &wg[(DDIM + 4 * j + q) * NEXP];
#pragma unroll
            for (int e = 0; e < NEXP; ++e) acc[e] += c[q] * wrow[e];
        }
    }

    // top-2 (first occurrence wins on ties, matching top_k)
    int i0 = 0; float l0 = acc[0];
#pragma unroll
    for (int e = 1; e < NEXP; ++e) if (acc[e] > l0) { l0 = acc[e]; i0 = e; }
    int i1 = -1; float l1 = -3.4e38f;
#pragma unroll
    for (int e = 0; e < NEXP; ++e) if (e != i0 && acc[e] > l1) { l1 = acc[e]; i1 = e; }

    float t = expf(l1 - l0);                   // stable 2-way softmax
    float inv = 1.f / (1.f + t);

    top_idx[row * 2 + 0] = i0;  top_idx[row * 2 + 1] = i1;
    top_gate[row * 2 + 0] = inv; top_gate[row * 2 + 1] = t * inv;
    atomicAdd(&counts[i0], 1);
    atomicAdd(&counts[i1], 1);
}

// ================================================================ scan (32-aligned expert segments)
__global__ void moe_scan_kernel(const int* __restrict__ counts,
                                int* __restrict__ off, int* __restrict__ cursor)
{
    if (threadIdx.x == 0) {
        int a = 0;
        for (int e = 0; e < NEXP; ++e) {
            off[e] = a; cursor[e] = a;
            a += ((counts[e] + 31) >> 5) << 5;
        }
        off[NEXP] = a;
    }
}

// ================================================================ placement
__global__ __launch_bounds__(256) void moe_place_kernel(
    const int* __restrict__ top_idx, const float* __restrict__ top_gate,
    int* __restrict__ cursor, int* __restrict__ slot_row, float* __restrict__ slot_gate)
{
    const int row = blockIdx.x * 256 + threadIdx.x;
#pragma unroll
    for (int k = 0; k < TOPK; ++k) {
        int e = top_idx[row * 2 + k];
        float g = top_gate[row * 2 + k];
        int pos = atomicAdd(&cursor[e], 1);
        slot_row[pos] = row;
        slot_gate[pos] = g;
    }
}

// ================================================================ f32->f16 convert + weight transpose
//  xh[i]        = x[i]
//  w1t[e][n][k] = w1[e][k][n]   (n<HDIM, k<DDIM)  -> GEMM1 B cols contiguous in K
//  w2t[e][n][k] = w2[e][k][n]   (n<DDIM, k<HDIM)  -> GEMM2 B cols contiguous in K
__global__ __launch_bounds__(256) void moe_cvt_kernel(
    const float* __restrict__ x,
    const float* __restrict__ w1, const float* __restrict__ w2,
    half_t* __restrict__ xh, half_t* __restrict__ w1t, half_t* __restrict__ w2t)
{
    int i = blockIdx.x * 256 + threadIdx.x;
    if (i < XELEM) xh[i] = (half_t)x[i];
    if (i < WELEM) {
        int e = i / (DDIM * HDIM), r = i % (DDIM * HDIM);
        { int n = r / DDIM, k = r - n * DDIM;              // w1t index r = n*DDIM + k
          w1t[i] = (half_t)w1[(size_t)(e * DDIM + k) * HDIM + n]; }
        { int n = r / HDIM, k = r - n * HDIM;              // w2t index r = n*HDIM + k
          w2t[i] = (half_t)w2[(size_t)(e * HDIM + k) * DDIM + n]; }
    }
}

// ================================================================ fragment builders (ISA §7.12.2 layouts)
__device__ inline v16h build_a_frag(const half_t* rowp, int kbase, int hi)
{
    v16h a;
#pragma unroll
    for (int i = 0; i < 4; ++i) {
        h2 p = *(const h2*)(rowp + kbase + hi * 8 + 2 * i);       // K = base..base+7
        a[2 * i] = p[0]; a[2 * i + 1] = p[1];
    }
#pragma unroll
    for (int i = 0; i < 4; ++i) {
        h2 p = *(const h2*)(rowp + kbase + 16 + hi * 8 + 2 * i);  // K = base+16..base+23
        a[8 + 2 * i] = p[0]; a[9 + 2 * i] = p[1];
    }
    return a;
}

__device__ inline v16h build_b_frag(const half_t* colp, int kbase, int hi)
{
    v16h b;
#pragma unroll
    for (int i = 0; i < 8; ++i) {
        h2 p = *(const h2*)(colp + kbase + (i >> 2) * 16 + hi * 8 + 2 * (i & 3));  // K = base+8h..+7, base+16+8h..+7
        b[2 * i] = p[0]; b[2 * i + 1] = p[1];
    }
    return b;
}

// ================================================================ main expert kernel
__global__ __launch_bounds__(256) void moe_expert_kernel(
    const half_t* __restrict__ xh,
    const half_t* __restrict__ w1t, const half_t* __restrict__ w2t,
    const float* __restrict__ b1, const float* __restrict__ b2,
    const int* __restrict__ off,
    const int* __restrict__ slot_row, const float* __restrict__ slot_gate,
    float* __restrict__ out)
{
    __shared__ __align__(16) half_t ldsX[MT * LDX];   // 12800 B
    __shared__ __align__(16) half_t ldsB[64 * LDB];   // 25600 B
    __shared__ __align__(16) half_t ldsH[MT * LDH];   // 24704 B
    __shared__ int   rowsS[MT];
    __shared__ float gatesS[MT];                      // total 63360 B

    const int t0 = blockIdx.x * MT;
    if (t0 >= off[NEXP]) return;                 // uniform early exit (before barriers)

    int e = 0;
    while (t0 >= off[e + 1]) ++e;                // tiles never straddle experts (32-aligned)

    const int tid  = threadIdx.x;
    const int wv   = tid >> 5;
    const int lane = tid & 31;
    const int hi   = lane >> 4;
    const int lm   = lane & 15;
    const int mI   = wv & 1;                     // wave's 16-row subtile (0/1)
    const int nl   = wv >> 1;                    // wave's 16-col subtile within 64-col chunk

    if (tid < MT) {
        rowsS[tid]  = slot_row[t0 + tid];
        gatesS[tid] = slot_gate[t0 + tid];
    }
    __syncthreads();

    // ---- gather 32 token rows (f16) via async DMA: 32 rows x 24 b128 segments
#pragma unroll
    for (int s = tid; s < MT * 24; s += 256) {
        int r = s / 24, sg = s - r * 24;
        int row = rowsS[r]; if (row < 0) row = 0;
        async_copy16(xh + (size_t)row * DDIM + sg * 8, &ldsX[r * LDX + sg * 8]);
    }
    async_wait0();
    __syncthreads();

    // ---- cache A fragments for GEMM1 (K = 192 -> 6 fragments of 32)
    v16h aF[6];
    {
        const half_t* rowp = &ldsX[(mI * 16 + lm) * LDX];
#pragma unroll
        for (int kb = 0; kb < 6; ++kb) aF[kb] = build_a_frag(rowp, kb * 32, hi);
    }

    const size_t eW = (size_t)e * DDIM * HDIM;

    // ================= GEMM1: H = gelu(X @ W1 + b1), 6 chunks of 64 cols
    for (int nc = 0; nc < 6; ++nc) {
        const int cb = nc * 64;
        // stage W1^T rows [cb..cb+64) x K=192 via async DMA: 64 x 24 b128 segments
#pragma unroll
        for (int s = tid; s < 64 * 24; s += 256) {
            int n = s / 24, sg = s - n * 24;
            async_copy16(w1t + eW + (size_t)(cb + n) * DDIM + sg * 8,
                         &ldsB[n * LDB + sg * 8]);
        }
        async_wait0();
        __syncthreads();

        v8f acc;
#pragma unroll
        for (int r = 0; r < 8; ++r) acc[r] = 0.f;

        const half_t* colp = &ldsB[(nl * 16 + lm) * LDB];
#pragma unroll
        for (int kb = 0; kb < 6; ++kb) {
            v16h bF = build_b_frag(colp, kb * 32, hi);
            acc = __builtin_amdgcn_wmma_f32_16x16x32_f16(
                false, aF[kb], false, bF, (short)0, acc, false, false);
        }

        // gelu (exact erf) -> f16 H tile
        const int nn = cb + nl * 16 + lm;
        const float bv = b1[e * HDIM + nn];
#pragma unroll
        for (int r = 0; r < 8; ++r) {
            int mm = mI * 16 + r + 8 * hi;
            float v = acc[r] + bv;
            float g = 0.5f * v * (1.0f + erff(v * 0.70710678118654752f));
            ldsH[mm * LDH + nn] = (half_t)g;
        }
        __syncthreads();
    }

    // ================= GEMM2: Y = H @ W2 + b2, 3 chunks of 64 cols, K=384 in 2 halves
    for (int nc2 = 0; nc2 < 3; ++nc2) {
        const int cb2 = nc2 * 64;
        v8f acc;
#pragma unroll
        for (int r = 0; r < 8; ++r) acc[r] = 0.f;

        for (int kc = 0; kc < 2; ++kc) {
            // stage W2^T rows [cb2..cb2+64) x K-half [kc*192..+192) via async DMA
#pragma unroll
            for (int s = tid; s < 64 * 24; s += 256) {
                int n = s / 24, sg = s - n * 24;
                async_copy16(w2t + eW + (size_t)(cb2 + n) * HDIM + kc * 192 + sg * 8,
                             &ldsB[n * LDB + sg * 8]);
            }
            async_wait0();
            __syncthreads();

            const half_t* rowp = &ldsH[(mI * 16 + lm) * LDH + kc * 192];
            const half_t* colp = &ldsB[(nl * 16 + lm) * LDB];
#pragma unroll
            for (int kb = 0; kb < 6; ++kb) {
                v16h aG = build_a_frag(rowp, kb * 32, hi);
                v16h bF = build_b_frag(colp, kb * 32, hi);
                acc = __builtin_amdgcn_wmma_f32_16x16x32_f16(
                    false, aG, false, bF, (short)0, acc, false, false);
            }
            __syncthreads();
        }

        // epilogue: combined[row] += gate * exp(y)   (two commutative f32 adds per cell)
        const int nn = cb2 + nl * 16 + lm;
        const float bv = b2[e * DDIM + nn];
#pragma unroll
        for (int r = 0; r < 8; ++r) {
            int mm = mI * 16 + r + 8 * hi;
            int row = rowsS[mm];
            if (row >= 0) {
                float y = acc[r] + bv;
                float contrib = gatesS[mm] * __expf(y);
                unsafeAtomicAdd(&out[(size_t)row * DDIM + nn], contrib);
            }
        }
    }
}

// ================================================================ finalize: log(combined==0 ? EPS : combined)
__global__ __launch_bounds__(256) void moe_final_kernel(float* __restrict__ out)
{
    int i = blockIdx.x * 256 + threadIdx.x;
    float4* p = (float4*)out;
    float4 v = p[i];
    v.x = logf(v.x == 0.f ? EPSV : v.x);
    v.y = logf(v.y == 0.f ? EPSV : v.y);
    v.z = logf(v.z == 0.f ? EPSV : v.z);
    v.w = logf(v.w == 0.f ? EPSV : v.w);
    p[i] = v;
}

// ================================================================ launch
extern "C" void kernel_launch(void* const* d_in, const int* in_sizes, int n_in,
                              void* d_out, int out_size, void* d_ws, size_t ws_size,
                              hipStream_t stream)
{
    const float* x      = (const float*)d_in[0];
    const float* emb    = (const float*)d_in[1];
    const float* w_gate = (const float*)d_in[2];
    const float* w1     = (const float*)d_in[3];
    const float* b1     = (const float*)d_in[4];
    const float* w2     = (const float*)d_in[5];
    const float* b2     = (const float*)d_in[6];
    float* out = (float*)d_out;

    char* ws = (char*)d_ws;
    int*    counts    = (int*)(ws + OFF_COUNTS);
    int*    offsets   = (int*)(ws + OFF_OFFSETS);
    int*    cursor    = (int*)(ws + OFF_CURSOR);
    int*    top_idx   = (int*)(ws + OFF_TOPIDX);
    float*  top_gate  = (float*)(ws + OFF_TOPGATE);
    int*    slot_row  = (int*)(ws + OFF_SLOTROW);
    float*  slot_gate = (float*)(ws + OFF_SLOTGATE);
    half_t* w1t       = (half_t*)(ws + OFF_W1T);
    half_t* w2t       = (half_t*)(ws + OFF_W2T);
    half_t* xh        = (half_t*)(ws + OFF_XH);

    // accumulate in d_out -> must zero every call (graph-safe async memsets)
    (void)hipMemsetAsync(d_out, 0, (size_t)NTOK * DDIM * sizeof(float), stream);
    (void)hipMemsetAsync(counts, 0, NEXP * sizeof(int), stream);
    (void)hipMemsetAsync(slot_row, 0xFF, (size_t)SLOT_CAP * sizeof(int), stream);  // -1 pad rows

    moe_gate_kernel<<<NTOK / 256, 256, 0, stream>>>(x, emb, w_gate, top_idx, top_gate, counts);
    moe_scan_kernel<<<1, 32, 0, stream>>>(counts, offsets, cursor);
    moe_place_kernel<<<NTOK / 256, 256, 0, stream>>>(top_idx, top_gate, cursor, slot_row, slot_gate);
    moe_cvt_kernel<<<XELEM / 256, 256, 0, stream>>>(x, w1, w2, xh, w1t, w2t);
    moe_expert_kernel<<<MAX_TILES, 256, 0, stream>>>(xh, w1t, w2t, b1, b2, offsets,
                                                     slot_row, slot_gate, out);
    moe_final_kernel<<<(NTOK * DDIM) / (256 * 4), 256, 0, stream>>>(out);
}
